// RNNAttentionModel_80814104641541
// MI455X (gfx1250) — matrix operand order repaired and checked by reading, not compiled, not run on hardware
//
#include <hip/hip_runtime.h>
#include <hip/hip_bf16.h>

// Problem constants (reference: T, B, H, L, V)
#define T_LEN  256
#define BATCH  16
#define HID    2048
#define NLAYER 3
#define VOCAB  32000
#define GATES  (4 * HID)        // 8192
#define TBR    (T_LEN * BATCH)  // 4096 rows

typedef __bf16 bf16_t;
typedef __attribute__((ext_vector_type(16))) __bf16 v16bf;
typedef __attribute__((ext_vector_type(8)))  __bf16 v8bf;
typedef __attribute__((ext_vector_type(8)))  float  v8f;

__device__ __forceinline__ float sigf(float x) { return 1.0f / (1.0f + __expf(-x)); }

// ---------------------------------------------------------------------------
// WMMA fragment loaders for V_WMMA_F32_16X16X32_BF16 (wave32).
// A (16xK-slice, row-major, lda elems): lane<16 holds row=lane,
// K {0..15}; lane>=16 holds row=lane-16, K {16..31}. Two 16B loads.
// B is Bw^T where Bw is [N,K] row-major: lane holds column n=lane&15,
// K range (lane>>4)*16 .. +15  -> one contiguous 32B load along K of Bw.
// ---------------------------------------------------------------------------
__device__ __forceinline__ v16bf load_a_frag(const bf16_t* base, int lda, int lane) {
  const int hi = lane >> 4;
  const bf16_t* p = base + (long long)(lane & 15) * lda + hi * 16;
  v8bf x0 = *(const v8bf*)(p);
  v8bf x1 = *(const v8bf*)(p + 8);
  v16bf r;
#pragma unroll
  for (int i = 0; i < 8; ++i) { r[i] = x0[i]; r[i + 8] = x1[i]; }
  return r;
}

__device__ __forceinline__ v16bf load_b_frag(const bf16_t* base, int ldb, int lane) {
  const bf16_t* p = base + (long long)(lane & 15) * ldb + (lane >> 4) * 16;
  return *(const v16bf*)p;
}

#define WMMA_BF16(a, b, c) \
  __builtin_amdgcn_wmma_f32_16x16x32_bf16(false, (a), false, (b), (short)0, (c), false, false)

// ---------------------------------------------------------------------------
// Generic batched NT GEMM:  C[z] = act( scale * (A[z] @ Bw[z]^T) + bias )
// Block = 128 threads = 4 waves; block tile 64(M) x 64(N); each wave: 16x64.
// Requires 64|M, 64|N, 32|K (true for every call site here).
// ---------------------------------------------------------------------------
__global__ __launch_bounds__(128) void gemm_nt_bf16(
    const bf16_t* __restrict__ A, const bf16_t* __restrict__ Bw,
    float* __restrict__ C, bf16_t* __restrict__ Cbf,
    const float* __restrict__ bias,
    int M, int N, int K, int lda, int ldb, int ldc,
    long long strideA, long long strideB, long long strideC,
    float scale, int act)
{
  const int z = blockIdx.z;
  A  += (long long)z * strideA;
  Bw += (long long)z * strideB;
  const long long coff = (long long)z * strideC;

  const int wave = threadIdx.x >> 5;
  const int lane = threadIdx.x & 31;
  const int m0 = blockIdx.y * 64 + wave * 16;
  const int n0 = blockIdx.x * 64;

  v8f acc[4] = {};
  const bf16_t* Abase = A + (long long)m0 * lda;

  for (int k0 = 0; k0 < K; k0 += 32) {
    v16bf a = load_a_frag(Abase + k0, lda, lane);
#pragma unroll
    for (int j = 0; j < 4; ++j) {
      v16bf b = load_b_frag(Bw + (long long)(n0 + j * 16) * ldb + k0, ldb, lane);
      acc[j] = WMMA_BF16(a, b, acc[j]);
    }
  }

  const int hi = lane >> 4, nl = lane & 15;
#pragma unroll
  for (int j = 0; j < 4; ++j) {
    const int n = n0 + j * 16 + nl;
    const float bb = bias ? bias[n] : 0.0f;
#pragma unroll
    for (int r = 0; r < 8; ++r) {
      const int m = m0 + r + 8 * hi;
      float v = acc[j][r] * scale + bb;
      if (act) v = tanhf(v);
      const long long off = coff + (long long)m * ldc + n;
      if (C)   C[off]   = v;
      if (Cbf) Cbf[off] = (bf16_t)v;
    }
  }
}

// ---------------------------------------------------------------------------
// One LSTM timestep: gates = xproj[t] + h @ Whh^T, fused nonlinearity.
// Grid = HID/64 blocks; block = 4 waves; wave owns a 16(batch)x16(n) patch and
// computes the i/f/g/o tiles at columns n, H+n, 2H+n, 3H+n so the cell update
// is register-local. h (16x2048 bf16, 64KB) is staged in LDS, shared by waves.
// Whh (32 MB bf16/layer) stays L2-resident across the 256 steps.
// ---------------------------------------------------------------------------
__global__ __launch_bounds__(128) void lstm_step_kernel(
    const bf16_t* __restrict__ h_in,   // [16, HID] bf16
    bf16_t* __restrict__ h_out,        // [16, HID] bf16 (ping-pong)
    bf16_t* __restrict__ x_out,        // &layer_out[t*B*H], [16, HID]
    float*  __restrict__ cst,          // [16, HID] fp32 cell state
    const float* __restrict__ xproj,   // [TBR, GATES]
    const bf16_t* __restrict__ Whh,    // [GATES, HID] bf16
    int t)
{
  __shared__ bf16_t hsh[16 * HID];  // 64 KB of the 320 KB WGP LDS
  {
    const uint4* src = (const uint4*)h_in;
    uint4* dst = (uint4*)hsh;
    for (int i = threadIdx.x; i < (16 * HID) / 8; i += 128) dst[i] = src[i];
  }
  __syncthreads();

  const int wave = threadIdx.x >> 5;
  const int lane = threadIdx.x & 31;
  const int n0 = blockIdx.x * 64 + wave * 16;

  v8f acc[4] = {};
  for (int k0 = 0; k0 < HID; k0 += 32) {
    v16bf a = load_a_frag(hsh + k0, HID, lane);  // ds_read path
#pragma unroll
    for (int g = 0; g < 4; ++g) {
      v16bf b = load_b_frag(Whh + (long long)(g * HID + n0) * HID + k0, HID, lane);
      acc[g] = WMMA_BF16(a, b, acc[g]);
    }
  }

  const int hi = lane >> 4, nl = lane & 15;
  const int n = n0 + nl;
#pragma unroll
  for (int r = 0; r < 8; ++r) {
    const int b = r + 8 * hi;
    const long long mrow = (long long)(t * BATCH + b) * GATES;
    const float gi = acc[0][r] + xproj[mrow + 0 * HID + n];
    const float gf = acc[1][r] + xproj[mrow + 1 * HID + n];
    const float gg = acc[2][r] + xproj[mrow + 2 * HID + n];
    const float go = acc[3][r] + xproj[mrow + 3 * HID + n];
    const int ci = b * HID + n;
    const float cn = sigf(gf) * cst[ci] + sigf(gi) * tanhf(gg);
    const float hn = sigf(go) * tanhf(cn);
    cst[ci] = cn;
    h_out[ci] = (bf16_t)hn;
    x_out[ci] = (bf16_t)hn;
  }
}

// ---------------------------------------------------------------------------
// Elementwise / reduction helpers
// ---------------------------------------------------------------------------
__global__ __launch_bounds__(256) void f32_to_bf16(const float* __restrict__ s,
                                                   bf16_t* __restrict__ d, size_t n) {
  size_t i = (size_t)blockIdx.x * 256 + threadIdx.x;
  if (i < n) d[i] = (bf16_t)s[i];
}

__global__ __launch_bounds__(256) void embed_gather(const int* __restrict__ ids,
                                                    const float* __restrict__ embW,
                                                    bf16_t* __restrict__ x, size_t n) {
  size_t i = (size_t)blockIdx.x * 256 + threadIdx.x;
  if (i < n) {
    const int tok = ids[i >> 11];  // HID = 2048
    x[i] = (bf16_t)embW[(size_t)tok * HID + (i & (HID - 1))];
  }
}

__global__ __launch_bounds__(256) void hc_init(const float* __restrict__ h0l,
                                               const float* __restrict__ c0l,
                                               bf16_t* __restrict__ hbf,
                                               float* __restrict__ c) {
  int i = blockIdx.x * 256 + threadIdx.x;
  if (i < BATCH * HID) { hbf[i] = (bf16_t)h0l[i]; c[i] = c0l[i]; }
}

// hsT[b][h][t] = hs[t][b][h]  (makes ctx GEMM an NT GEMM, contiguous in K=t)
__global__ __launch_bounds__(256) void transpose_hs(const bf16_t* __restrict__ hs,
                                                    bf16_t* __restrict__ hsT, size_t n) {
  size_t i = (size_t)blockIdx.x * 256 + threadIdx.x;
  if (i < n) {
    const int t = (int)(i / (BATCH * HID));
    const int rem = (int)(i % (BATCH * HID));
    const int b = rem / HID, h = rem % HID;
    hsT[((long long)b * HID + h) * T_LEN + t] = hs[i];
  }
}

// causal softmax over s for each (t, b) row; writes bf16 attn[b][t][s]
__global__ __launch_bounds__(256) void softmax_causal(const float* __restrict__ scores,
                                                      bf16_t* __restrict__ attn) {
  __shared__ float red[256];
  const int t = blockIdx.x, b = blockIdx.y, s = threadIdx.x;
  const long long off = ((long long)b * T_LEN + t) * T_LEN + s;
  const bool valid = (s <= t);
  const float v = valid ? scores[off] : -3.0e38f;
  red[s] = v; __syncthreads();
#pragma unroll
  for (int w = 128; w >= 1; w >>= 1) {
    if (s < w) red[s] = fmaxf(red[s], red[s + w]);
    __syncthreads();
  }
  const float mx = red[0];
  __syncthreads();
  const float e = valid ? __expf(v - mx) : 0.0f;
  red[s] = e; __syncthreads();
#pragma unroll
  for (int w = 128; w >= 1; w >>= 1) {
    if (s < w) red[s] += red[s + w];
    __syncthreads();
  }
  attn[off] = (bf16_t)(e / red[0]);
}

// cat[m][0:H] = hs row m; cat[m][H:2H] = ctx[b][t][:]   (m = t*B + b)
__global__ __launch_bounds__(256) void concat_hs_ctx(const bf16_t* __restrict__ hs,
                                                     const bf16_t* __restrict__ ctx,
                                                     bf16_t* __restrict__ cat, size_t n) {
  size_t i = (size_t)blockIdx.x * 256 + threadIdx.x;
  if (i < n) {
    const int m = (int)(i / (2 * HID));
    const int col = (int)(i % (2 * HID));
    const int t = m / BATCH, b = m % BATCH;
    cat[i] = (col < HID)
                 ? hs[(long long)m * HID + col]
                 : ctx[((long long)b * T_LEN + t) * HID + (col - HID)];
  }
}

// ---------------------------------------------------------------------------
extern "C" void kernel_launch(void* const* d_in, const int* in_sizes, int n_in,
                              void* d_out, int out_size, void* d_ws, size_t ws_size,
                              hipStream_t stream) {
  (void)in_sizes; (void)n_in; (void)out_size; (void)ws_size;

  const int*   ids  = (const int*)  d_in[0];
  const float* h0   = (const float*)d_in[1];
  const float* c0   = (const float*)d_in[2];
  const float* embW = (const float*)d_in[3];
  const float* Wih  = (const float*)d_in[4];
  const float* Whh  = (const float*)d_in[5];
  const float* bvec = (const float*)d_in[6];
  const float* Watt = (const float*)d_in[7];
  const float* batt = (const float*)d_in[8];
  const float* decW = (const float*)d_in[9];
  const float* decb = (const float*)d_in[10];
  float* out = (float*)d_out;

  // Workspace carve-out (256B aligned slices)
  char* p = (char*)d_ws;
  auto take = [&p](size_t bytes) -> char* {
    char* r = p; p += (bytes + 255) & ~(size_t)255; return r;
  };
  const size_t nW = (size_t)NLAYER * GATES * HID;
  bf16_t* WihB  = (bf16_t*)take(nW * 2);
  bf16_t* WhhB  = (bf16_t*)take(nW * 2);
  bf16_t* WattB = (bf16_t*)take((size_t)HID * 2 * HID * 2);
  bf16_t* decWB = (bf16_t*)take((size_t)VOCAB * HID * 2);
  bf16_t* xA    = (bf16_t*)take((size_t)TBR * HID * 2);
  bf16_t* xB    = (bf16_t*)take((size_t)TBR * HID * 2);
  float*  xproj = (float*) take((size_t)TBR * GATES * 4);
  float*  cst   = (float*) take((size_t)BATCH * HID * 4);
  bf16_t* hp0   = (bf16_t*)take((size_t)BATCH * HID * 2);
  bf16_t* hp1   = (bf16_t*)take((size_t)BATCH * HID * 2);
  float*  sc    = (float*) take((size_t)BATCH * T_LEN * T_LEN * 4);
  bf16_t* attn  = (bf16_t*)take((size_t)BATCH * T_LEN * T_LEN * 2);
  bf16_t* hsT   = (bf16_t*)take((size_t)BATCH * HID * T_LEN * 2);
  bf16_t* ctxB  = (bf16_t*)take((size_t)BATCH * T_LEN * HID * 2);
  bf16_t* catB  = (bf16_t*)take((size_t)TBR * 2 * HID * 2);
  bf16_t* outB  = (bf16_t*)take((size_t)TBR * HID * 2);

  auto cg = [](size_t n) { return dim3((unsigned)((n + 255) / 256)); };

  // 1) One-time bf16 weight conversion (stays L2/HBM resident thereafter)
  f32_to_bf16<<<cg(nW), 256, 0, stream>>>(Wih, WihB, nW);
  f32_to_bf16<<<cg(nW), 256, 0, stream>>>(Whh, WhhB, nW);
  f32_to_bf16<<<cg((size_t)HID * 2 * HID), 256, 0, stream>>>(Watt, WattB, (size_t)HID * 2 * HID);
  f32_to_bf16<<<cg((size_t)VOCAB * HID), 256, 0, stream>>>(decW, decWB, (size_t)VOCAB * HID);

  // 2) Embedding gather -> bf16
  embed_gather<<<cg((size_t)TBR * HID), 256, 0, stream>>>(ids, embW, xA, (size_t)TBR * HID);

  // 3) Three LSTM layers
  bf16_t* xin = xA;
  bf16_t* xout = xB;
  for (int l = 0; l < NLAYER; ++l) {
    hc_init<<<cg((size_t)BATCH * HID), 256, 0, stream>>>(
        h0 + (size_t)l * BATCH * HID, c0 + (size_t)l * BATCH * HID, hp0, cst);

    // xproj = x @ Wih_l^T + b_l   [4096 x 8192], fp32 accumulate
    gemm_nt_bf16<<<dim3(GATES / 64, TBR / 64, 1), 128, 0, stream>>>(
        xin, WihB + (size_t)l * GATES * HID, xproj, nullptr, bvec + (size_t)l * GATES,
        TBR, GATES, HID, HID, HID, GATES, 0, 0, 0, 1.0f, 0);

    // 256 sequential recurrent steps (stream-ordered launches)
    for (int t = 0; t < T_LEN; ++t) {
      lstm_step_kernel<<<HID / 64, 128, 0, stream>>>(
          (t & 1) ? hp1 : hp0, (t & 1) ? hp0 : hp1,
          xout + (size_t)t * BATCH * HID, cst, xproj,
          WhhB + (size_t)l * GATES * HID, t);
    }
    bf16_t* tmp = xin; xin = xout; xout = tmp;
  }
  bf16_t* hs = xin;  // last-layer hidden states [T, B, H] bf16

  // 4) Causal self-attention over the hidden-state history
  transpose_hs<<<cg((size_t)TBR * HID), 256, 0, stream>>>(hs, hsT, (size_t)TBR * HID);

  // scores[b] = (hs_b @ hs_b^T) / sqrt(H)   (batched over b)
  gemm_nt_bf16<<<dim3(T_LEN / 64, T_LEN / 64, BATCH), 128, 0, stream>>>(
      hs, hs, sc, nullptr, nullptr,
      T_LEN, T_LEN, HID, BATCH * HID, BATCH * HID, T_LEN,
      (long long)HID, (long long)HID, (long long)T_LEN * T_LEN,
      0.022097086912079608f /* 1/sqrt(2048) */, 0);

  softmax_causal<<<dim3(T_LEN, BATCH), 256, 0, stream>>>(sc, attn);

  // ctx[b] = attn_b @ hs_b  via NT with hsT  (K = T)
  gemm_nt_bf16<<<dim3(HID / 64, T_LEN / 64, BATCH), 128, 0, stream>>>(
      attn, hsT, nullptr, ctxB, nullptr,
      T_LEN, HID, T_LEN, T_LEN, T_LEN, HID,
      (long long)T_LEN * T_LEN, (long long)HID * T_LEN, (long long)T_LEN * HID,
      1.0f, 0);

  // 5) out = tanh(cat(hs, ctx) @ W_att^T + b_att)  -> bf16 for decoder
  concat_hs_ctx<<<cg((size_t)TBR * 2 * HID), 256, 0, stream>>>(
      hs, ctxB, catB, (size_t)TBR * 2 * HID);
  gemm_nt_bf16<<<dim3(HID / 64, TBR / 64, 1), 128, 0, stream>>>(
      catB, WattB, nullptr, outB, batt,
      TBR, HID, 2 * HID, 2 * HID, 2 * HID, HID, 0, 0, 0, 1.0f, 1 /*tanh*/);

  // 6) decoded = out @ dec_W^T + dec_b  -> fp32 d_out [4096 x 32000]
  gemm_nt_bf16<<<dim3(VOCAB / 64, TBR / 64, 1), 128, 0, stream>>>(
      outB, decWB, out, nullptr, decb,
      TBR, VOCAB, HID, HID, HID, VOCAB, 0, 0, 0, 1.0f, 0);
}
